// FLAPaTH_5351529251282
// MI455X (gfx1250) — matrix operand order repaired and checked by reading, not compiled, not run on hardware
//
#include <hip/hip_runtime.h>
#include <stdint.h>
#include <stddef.h>

// PaTH attention forward for MI455X (gfx1250, wave32, WMMA).
// All heavy GEMMs use v_wmma_f32_16x16x32_bf16 (bf16 in, fp32 accumulate).
// The O(T^3) UT-transform triangular solve is factorized (rank-D) into 16
// dependent chunk steps of [T x 64] @ [64 x 64] WMMA GEMMs with a running
// fp32 accumulator, so total matrix work is ~45 GFLOP vs ~0.2 GB of traffic:
// hard compute-bound -> WMMA everywhere.
//
// Round-2 changes: inner MMA loops restructured (1 A-frag live at a time,
// #pragma unroll 1 on K-loops, __launch_bounds__(32,1)) to eliminate the
// scratch spills seen in round 1, plus global_prefetch of the next K-tile.

#define T_SEQ  1024
#define DMODEL 1024
#define NHEAD  16
#define HDIM   64
#define NCHUNK 16
#define RANK_  32

typedef __attribute__((ext_vector_type(16))) __bf16    v16bf;
typedef __attribute__((ext_vector_type(8)))  float     v8f;
typedef __attribute__((ext_vector_type(4)))  float     v4f;
typedef __attribute__((ext_vector_type(8)))  uint16_t  u16x8;

union BF16Frag { v16bf v; u16x8 h[2]; uint16_t u[16]; };

static __device__ __forceinline__ uint16_t f2b(float f) {
  uint32_t u = __float_as_uint(f);
  uint32_t r = (u + 0x7fffu + ((u >> 16) & 1u)) >> 16;  // RNE
  return (uint16_t)r;
}
static __device__ __forceinline__ float b2f(uint16_t h) {
  return __uint_as_float(((uint32_t)h) << 16);
}

// ---------------- WMMA fragment loaders (CDNA5 16x16x32 bf16 layouts) -------
// A (16x32, MxK): lane&15 = row M; lane[4] selects K-chunks {0..7,16..23} vs
// {8..15,24..31}; each chunk is 8 contiguous bf16 = one 16B load.
static __device__ __forceinline__ v16bf load_a_bf16(const uint16_t* A, int lda, int k0) {
  const int lane = threadIdx.x & 31;
  const uint16_t* p = A + (size_t)(lane & 15) * lda + k0 + ((lane >> 4) * 8);
  BF16Frag f;
  f.h[0] = *(const u16x8*)p;
  f.h[1] = *(const u16x8*)(p + 16);
  return f.v;
}
// Same A layout but source is fp32 (running accumulator), convert on the fly.
static __device__ __forceinline__ v16bf load_a_f32(const float* A, int lda, int k0) {
  const int lane = threadIdx.x & 31;
  const float* p = A + (size_t)(lane & 15) * lda + k0 + ((lane >> 4) * 8);
  v4f a0 = *(const v4f*)p;
  v4f a1 = *(const v4f*)(p + 4);
  v4f b0 = *(const v4f*)(p + 16);
  v4f b1 = *(const v4f*)(p + 20);
  BF16Frag f;
#pragma unroll
  for (int i = 0; i < 4; ++i) {
    f.u[i]      = f2b(a0[i]);
    f.u[4 + i]  = f2b(a1[i]);
    f.u[8 + i]  = f2b(b0[i]);
    f.u[12 + i] = f2b(b1[i]);
  }
  return f.v;
}
// B (32x16, KxN), sourced from B^T stored row-major [N x K]: lane&15 = col N,
// lane[4] selects K-chunks {0..7,16..23} vs {8..15,24..31} -> two 16B loads.
static __device__ __forceinline__ v16bf load_b_bf16(const uint16_t* Bt, int ldb, int k0) {
  const int lane = threadIdx.x & 31;
  const uint16_t* p = Bt + (size_t)(lane & 15) * ldb + k0 + ((lane >> 4) * 8);
  BF16Frag f;
  f.h[0] = *(const u16x8*)p;
  f.h[1] = *(const u16x8*)(p + 16);
  return f.v;
}

static __device__ __forceinline__ void zero_acc(v8f acc[4][4]) {
#pragma unroll
  for (int m = 0; m < 4; ++m)
#pragma unroll
    for (int n = 0; n < 4; ++n)
#pragma unroll
      for (int i = 0; i < 8; ++i) acc[m][n][i] = 0.f;
}

#define WMMA_BF16(a, b, c) \
  __builtin_amdgcn_wmma_f32_16x16x32_bf16(false, (a), false, (b), (short)0, (c), false, false)

// One wave computes a 64x64 fp32 tile: C += A[64xK] * (Bt[64xK])^T.
// Only 4 B-frags + 1 A-frag live at a time; K-loop kept rolled to avoid the
// register-pressure spills that multi-iteration pipelining caused.
static __device__ __forceinline__ void mma64_bf16(v8f acc[4][4], const uint16_t* A, int lda,
                                                  const uint16_t* Bt, int ldb, int K,
                                                  bool prefetch) {
#pragma unroll 1
  for (int k0 = 0; k0 < K; k0 += 32) {
    if (prefetch && (k0 + 32 < K)) {
      __builtin_prefetch(A + (size_t)(threadIdx.x & 31) * lda + k0 + 32, 0, 0);
      __builtin_prefetch(Bt + (size_t)(threadIdx.x & 31) * ldb + k0 + 32, 0, 0);
    }
    v16bf b0 = load_b_bf16(Bt, ldb, k0);
    v16bf b1 = load_b_bf16(Bt + (size_t)16 * ldb, ldb, k0);
    v16bf b2 = load_b_bf16(Bt + (size_t)32 * ldb, ldb, k0);
    v16bf b3 = load_b_bf16(Bt + (size_t)48 * ldb, ldb, k0);
#pragma unroll
    for (int m = 0; m < 4; ++m) {
      v16bf a = load_a_bf16(A + (size_t)(m * 16) * lda, lda, k0);
      acc[m][0] = WMMA_BF16(a, b0, acc[m][0]);
      acc[m][1] = WMMA_BF16(a, b1, acc[m][1]);
      acc[m][2] = WMMA_BF16(a, b2, acc[m][2]);
      acc[m][3] = WMMA_BF16(a, b3, acc[m][3]);
    }
  }
}
static __device__ __forceinline__ void mma64_f32a(v8f acc[4][4], const float* A, int lda,
                                                  const uint16_t* Bt, int ldb, int K) {
#pragma unroll 1
  for (int k0 = 0; k0 < K; k0 += 32) {
    v16bf b0 = load_b_bf16(Bt, ldb, k0);
    v16bf b1 = load_b_bf16(Bt + (size_t)16 * ldb, ldb, k0);
    v16bf b2 = load_b_bf16(Bt + (size_t)32 * ldb, ldb, k0);
    v16bf b3 = load_b_bf16(Bt + (size_t)48 * ldb, ldb, k0);
#pragma unroll
    for (int m = 0; m < 4; ++m) {
      v16bf a = load_a_f32(A + (size_t)(m * 16) * lda, lda, k0);
      acc[m][0] = WMMA_BF16(a, b0, acc[m][0]);
      acc[m][1] = WMMA_BF16(a, b1, acc[m][1]);
      acc[m][2] = WMMA_BF16(a, b2, acc[m][2]);
      acc[m][3] = WMMA_BF16(a, b3, acc[m][3]);
    }
  }
}

// --------------------------- generic batched GEMM ---------------------------
// C[M x N] = A[M x K] * B, with B given transposed (Bt row-major [N x K]).
// One wave per 64x64 output tile. OUTMODE: 0 = fp32, 1 = bf16, 2 = bf16
// transposed store (writes C^T, used to produce v^T directly).
template <int OUTMODE>
__global__ __launch_bounds__(32, 1) void gemm64_kernel(
    const uint16_t* __restrict__ A, int lda, long long sA,
    const uint16_t* __restrict__ Bt, int ldb, long long sB,
    void* __restrict__ Cv, int ldc, long long sC,
    int K, float scale) {
  const int n0 = blockIdx.x * 64;
  const int m0 = blockIdx.y * 64;
  const int bz = blockIdx.z;
  const uint16_t* Ab = A + (size_t)bz * sA + (size_t)m0 * lda;
  const uint16_t* Bb = Bt + (size_t)bz * sB + (size_t)n0 * ldb;
  v8f acc[4][4];
  zero_acc(acc);
  mma64_bf16(acc, Ab, lda, Bb, ldb, K, true);
  const int lane = threadIdx.x & 31;
  const int rb = (lane >> 4) * 8, cl = lane & 15;
  if (OUTMODE == 0) {
    float* C = (float*)Cv + (size_t)bz * sC;
#pragma unroll
    for (int m = 0; m < 4; ++m)
#pragma unroll
      for (int n = 0; n < 4; ++n)
#pragma unroll
        for (int r = 0; r < 8; ++r)
          C[(size_t)(m0 + m * 16 + rb + r) * ldc + n0 + n * 16 + cl] = acc[m][n][r] * scale;
  } else if (OUTMODE == 1) {
    uint16_t* C = (uint16_t*)Cv + (size_t)bz * sC;
#pragma unroll
    for (int m = 0; m < 4; ++m)
#pragma unroll
      for (int n = 0; n < 4; ++n)
#pragma unroll
        for (int r = 0; r < 8; ++r)
          C[(size_t)(m0 + m * 16 + rb + r) * ldc + n0 + n * 16 + cl] = f2b(acc[m][n][r] * scale);
  } else {
    uint16_t* C = (uint16_t*)Cv + (size_t)bz * sC;
#pragma unroll
    for (int m = 0; m < 4; ++m)
#pragma unroll
      for (int n = 0; n < 4; ++n)
#pragma unroll
        for (int r = 0; r < 8; ++r)
          C[(size_t)(n0 + n * 16 + cl) * ldc + m0 + m * 16 + rb + r] = f2b(acc[m][n][r] * scale);
  }
}

// --------------------------- small / setup kernels ---------------------------
__global__ void cast_bf16_kernel(const float* __restrict__ in, uint16_t* __restrict__ out, int n) {
  int i = blockIdx.x * blockDim.x + threadIdx.x;
  if (i < n) out[i] = f2b(in[i]);
}

// in: [rows x cols] fp32 -> out: [cols x rows] bf16 (i.e. out = in^T)
__global__ void transpose_cast_kernel(const float* __restrict__ in, uint16_t* __restrict__ out,
                                      int rows, int cols) {
  int i = blockIdx.x * blockDim.x + threadIdx.x;
  if (i >= rows * cols) return;
  int n = i / rows, k = i % rows;
  out[(size_t)n * rows + k] = f2b(in[(size_t)k * cols + n]);
}

// beta = 2*sigmoid(x @ Wb), [T x H]
__global__ void proj_beta_kernel(const float* __restrict__ x, const float* __restrict__ Wb,
                                 float* __restrict__ beta) {
  int idx = blockIdx.x * blockDim.x + threadIdx.x;  // T*H threads
  int t = idx / NHEAD, h = idx % NHEAD;
  float s = 0.f;
  const float* xr = x + (size_t)t * DMODEL;
  for (int d = 0; d < DMODEL; ++d) s += xr[d] * Wb[(size_t)d * NHEAD + h];
  beta[idx] = 2.f / (1.f + expf(-s));
}

// xw1 = x @ Ww1, [T x RANK]
__global__ void proj_xw1_kernel(const float* __restrict__ x, const float* __restrict__ Ww1,
                                float* __restrict__ xw1) {
  int idx = blockIdx.x * blockDim.x + threadIdx.x;  // T*RANK threads
  int t = idx / RANK_, r = idx % RANK_;
  float s = 0.f;
  const float* xr = x + (size_t)t * DMODEL;
  for (int d = 0; d < DMODEL; ++d) s += xr[d] * Ww1[(size_t)d * RANK_ + r];
  xw1[idx] = s;
}

// wlow = xw1 @ Ww2, [T x DM]
__global__ void wlow_kernel(const float* __restrict__ xw1, const float* __restrict__ Ww2,
                            float* __restrict__ wlow) {
  int idx = blockIdx.x * blockDim.x + threadIdx.x;  // T*DM threads
  int t = idx / DMODEL, c = idx % DMODEL;
  float s = 0.f;
#pragma unroll
  for (int r = 0; r < RANK_; ++r) s += xw1[(size_t)t * RANK_ + r] * Ww2[(size_t)r * DMODEL + c];
  wlow[idx] = s;
}

// Causal depthwise conv(3) + SiLU + per-head L2 norm; emits w (bf16),
// bw = beta*w (bf16, row-major) and bw^T (bf16, channel-major for B-frags).
__global__ void conv_norm_kernel(const float* __restrict__ wlow, const float* __restrict__ conv_w,
                                 const float* __restrict__ beta, uint16_t* __restrict__ w_bf,
                                 uint16_t* __restrict__ bw_bf, uint16_t* __restrict__ bwT_bf) {
  const int t = blockIdx.x, h = blockIdx.y;
  const int c = h * HDIM + threadIdx.x;
  float y = 0.f;
#pragma unroll
  for (int m = 0; m < 3; ++m) {
    int tt = t - 2 + m;
    if (tt >= 0) y += conv_w[(size_t)c * 3 + m] * wlow[(size_t)tt * DMODEL + c];
  }
  float sil = y / (1.f + expf(-y));
  __shared__ float red[HDIM];
  red[threadIdx.x] = sil * sil;
  __syncthreads();
  for (int s = HDIM / 2; s > 0; s >>= 1) {
    if ((int)threadIdx.x < s) red[threadIdx.x] += red[threadIdx.x + s];
    __syncthreads();
  }
  float wv = sil * rsqrtf(red[0] + 1e-6f);
  w_bf[(size_t)t * DMODEL + c] = f2b(wv);
  float bv = beta[(size_t)t * NHEAD + h] * wv;
  bw_bf[(size_t)t * DMODEL + c] = f2b(bv);
  bwT_bf[(size_t)c * T_SEQ + t] = f2b(bv);
}

// Per (head, chunk): Tinv = (I + strictlower(bw_J w_J^T))^-1 (stored transposed
// bf16 for B-frag use) and LkT = strictlower(bw_J k_J^T)^T (bf16).
__global__ void invert_kernel(const uint16_t* __restrict__ wb, const uint16_t* __restrict__ kb,
                              const uint16_t* __restrict__ bwb, uint16_t* __restrict__ TinvT,
                              uint16_t* __restrict__ LkT) {
  __shared__ float Ab[64][65];
  __shared__ float Xb[64][65];
  const int J = blockIdx.x, h = blockIdx.y;
  const int Jc = J * 64, c0 = h * HDIM;
  const int r = threadIdx.x;  // 64 threads

  float breg[HDIM];
  const uint16_t* bwr = bwb + (size_t)(Jc + r) * DMODEL + c0;
#pragma unroll
  for (int d = 0; d < HDIM; ++d) breg[d] = b2f(bwr[d]);
  for (int c = 0; c < 64; ++c) {
    const uint16_t* wc = wb + (size_t)(Jc + c) * DMODEL + c0;
    float s = 0.f;
#pragma unroll
    for (int d = 0; d < HDIM; ++d) s += breg[d] * b2f(wc[d]);
    Ab[r][c] = (r > c) ? s : (r == c ? 1.f : 0.f);
  }
  __syncthreads();
  // Forward substitution inversion of unit-lower-triangular A; thread = column.
  const int j = r;
  for (int i = 0; i < 64; ++i) {
    float s = (i == j) ? 1.f : 0.f;
    for (int t = 0; t < i; ++t) s -= Ab[i][t] * Xb[t][j];
    Xb[i][j] = s;
    __syncthreads();
  }
  uint16_t* outT = TinvT + (size_t)(h * NCHUNK + J) * 4096 + (size_t)j * 64;
  for (int s = 0; s < 64; ++s) outT[s] = f2b(Xb[s][j]);

  // LkT row j: Lk[s,j] = (s>j) ? bw_s . k_j : 0
  float kreg[HDIM];
  const uint16_t* kj = kb + (size_t)(Jc + j) * DMODEL + c0;
#pragma unroll
  for (int d = 0; d < HDIM; ++d) kreg[d] = b2f(kj[d]);
  uint16_t* outL = LkT + (size_t)(h * NCHUNK + J) * 4096 + (size_t)j * 64;
  for (int s = 0; s < 64; ++s) {
    float acc = 0.f;
    if (s > j) {
      const uint16_t* bws = bwb + (size_t)(Jc + s) * DMODEL + c0;
#pragma unroll
      for (int d = 0; d < HDIM; ++d) acc += b2f(bws[d]) * kreg[d];
    }
    outL[s] = f2b(acc);
  }
}

// ------------------------ chunked UT-transform step -------------------------
// For column chunk J (right-to-left): per (head, row-tile i0>=Jc), all 64x64:
//   R   = tril(q w_J^T) - Acc w_J^T
//   G_J = R Tinv_J
//   S[:,J] = (q k_J^T - Acc k_J^T - G_J Lk_J) * D^-0.5
//   Acc += G_J bw_J
__global__ __launch_bounds__(32, 1) void path_step_kernel(
    int J, const uint16_t* __restrict__ qb, const uint16_t* __restrict__ kb,
    const uint16_t* __restrict__ wb, const uint16_t* __restrict__ bwT,
    const uint16_t* __restrict__ TinvT, const uint16_t* __restrict__ LkT,
    float* __restrict__ Acc, float* __restrict__ S) {
  __shared__ __align__(16) float    Ff[64 * 64];
  __shared__ __align__(16) uint16_t Rs[64 * 64];
  __shared__ __align__(16) uint16_t Gs[64 * 64];

  const int h = blockIdx.y;
  const int it = J + blockIdx.x;
  const int i0 = it * 64;
  const int Jc = J * 64;
  const bool diag = (blockIdx.x == 0);

  const uint16_t* qA = qb + (size_t)i0 * DMODEL + h * HDIM;
  const uint16_t* wB = wb + (size_t)Jc * DMODEL + h * HDIM;
  const uint16_t* kB = kb + (size_t)Jc * DMODEL + h * HDIM;
  const uint16_t* TB = TinvT + (size_t)(h * NCHUNK + J) * 4096;
  const uint16_t* LB = LkT + (size_t)(h * NCHUNK + J) * 4096;
  const uint16_t* bwB = bwT + (size_t)(h * HDIM) * T_SEQ + Jc;
  float* AccT = Acc + ((size_t)h * T_SEQ + i0) * HDIM;

  const int lane = threadIdx.x & 31;
  const int rb = (lane >> 4) * 8, cl = lane & 15;
  v8f acc[4][4];

  // (1) Ff = q @ w_J^T
  zero_acc(acc);
  mma64_bf16(acc, qA, DMODEL, wB, DMODEL, HDIM, false);
#pragma unroll
  for (int m = 0; m < 4; ++m)
#pragma unroll
    for (int n = 0; n < 4; ++n)
#pragma unroll
      for (int r = 0; r < 8; ++r)
        Ff[(m * 16 + rb + r) * 64 + n * 16 + cl] = acc[m][n][r];
  __syncthreads();

  // (2) R = Ff - Acc @ w_J^T, causal-masked on the diagonal tile -> Rs (bf16)
  zero_acc(acc);
  mma64_f32a(acc, AccT, HDIM, wB, DMODEL, HDIM);
#pragma unroll
  for (int m = 0; m < 4; ++m)
#pragma unroll
    for (int n = 0; n < 4; ++n)
#pragma unroll
      for (int r = 0; r < 8; ++r) {
        int lr = m * 16 + rb + r, lc = n * 16 + cl;
        float v = Ff[lr * 64 + lc] - acc[m][n][r];
        if (diag && lr < lc) v = 0.f;
        Rs[lr * 64 + lc] = f2b(v);
      }
  __syncthreads();

  // (3) G = R @ Tinv_J -> Gs (bf16)
  zero_acc(acc);
  mma64_bf16(acc, Rs, 64, TB, 64, HDIM, false);
#pragma unroll
  for (int m = 0; m < 4; ++m)
#pragma unroll
    for (int n = 0; n < 4; ++n)
#pragma unroll
      for (int r = 0; r < 8; ++r)
        Gs[(m * 16 + rb + r) * 64 + n * 16 + cl] = f2b(acc[m][n][r]);
  __syncthreads();

  // (4) Ff = Acc @ k_J^T + G @ Lk_J
  zero_acc(acc);
  mma64_f32a(acc, AccT, HDIM, kB, DMODEL, HDIM);
  mma64_bf16(acc, Gs, 64, LB, 64, HDIM, false);
#pragma unroll
  for (int m = 0; m < 4; ++m)
#pragma unroll
    for (int n = 0; n < 4; ++n)
#pragma unroll
      for (int r = 0; r < 8; ++r)
        Ff[(m * 16 + rb + r) * 64 + n * 16 + cl] = acc[m][n][r];
  __syncthreads();

  // (5) S block = (q @ k_J^T - Ff) * D^-0.5
  zero_acc(acc);
  mma64_bf16(acc, qA, DMODEL, kB, DMODEL, HDIM, false);
  float* Srow = S + ((size_t)h * T_SEQ + i0) * T_SEQ + Jc;
#pragma unroll
  for (int m = 0; m < 4; ++m)
#pragma unroll
    for (int n = 0; n < 4; ++n)
#pragma unroll
      for (int r = 0; r < 8; ++r) {
        int lr = m * 16 + rb + r, lc = n * 16 + cl;
        Srow[(size_t)lr * T_SEQ + lc] = (acc[m][n][r] - Ff[lr * 64 + lc]) * 0.125f;
      }

  // (6) Acc += G @ bw_J
  zero_acc(acc);
  mma64_bf16(acc, Gs, 64, bwB, T_SEQ, HDIM, false);
#pragma unroll
  for (int m = 0; m < 4; ++m)
#pragma unroll
    for (int n = 0; n < 4; ++n)
#pragma unroll
      for (int r = 0; r < 8; ++r)
        AccT[(size_t)(m * 16 + rb + r) * HDIM + n * 16 + cl] += acc[m][n][r];
}

// Row softmax over j<=i; writes P in bf16 with explicit zeros for j>i.
__global__ void softmax_kernel(const float* __restrict__ S, uint16_t* __restrict__ P) {
  const int i = blockIdx.x, h = blockIdx.y;
  const float* row = S + ((size_t)h * T_SEQ + i) * T_SEQ;
  uint16_t* prow = P + ((size_t)h * T_SEQ + i) * T_SEQ;
  __shared__ float red[256];
  const int tid = threadIdx.x;
  float m = -1e30f;
  for (int j = tid; j <= i; j += 256) m = fmaxf(m, row[j]);
  red[tid] = m;
  __syncthreads();
  for (int s = 128; s > 0; s >>= 1) {
    if (tid < s) red[tid] = fmaxf(red[tid], red[tid + s]);
    __syncthreads();
  }
  m = red[0];
  __syncthreads();
  float sum = 0.f;
  for (int j = tid; j <= i; j += 256) sum += expf(row[j] - m);
  red[tid] = sum;
  __syncthreads();
  for (int s = 128; s > 0; s >>= 1) {
    if (tid < s) red[tid] += red[tid + s];
    __syncthreads();
  }
  float inv = 1.f / red[0];
  for (int j = tid; j < T_SEQ; j += 256) {
    float p = (j <= i) ? expf(row[j] - m) * inv : 0.f;
    prow[j] = f2b(p);
  }
}

// ------------------------------- host driver --------------------------------
extern "C" void kernel_launch(void* const* d_in, const int* in_sizes, int n_in,
                              void* d_out, int out_size, void* d_ws, size_t ws_size,
                              hipStream_t stream) {
  (void)in_sizes; (void)n_in; (void)out_size; (void)ws_size;
  const float* x     = (const float*)d_in[0];
  const float* Wq    = (const float*)d_in[1];
  const float* Wk    = (const float*)d_in[2];
  const float* Wv    = (const float*)d_in[3];
  const float* Ww1   = (const float*)d_in[4];
  const float* Ww2   = (const float*)d_in[5];
  const float* convw = (const float*)d_in[6];
  const float* Wb    = (const float*)d_in[7];
  const float* Wo    = (const float*)d_in[8];
  float* out = (float*)d_out;

  char* ws = (char*)d_ws;
  size_t off = 0;
  auto carve = [&](size_t bytes) -> void* {
    void* p = ws + off;
    off = (off + bytes + 255) & ~(size_t)255;
    return p;
  };
  uint16_t* x_bf   = (uint16_t*)carve((size_t)T_SEQ * DMODEL * 2);
  uint16_t* WqT    = (uint16_t*)carve((size_t)DMODEL * DMODEL * 2);
  uint16_t* WkT    = (uint16_t*)carve((size_t)DMODEL * DMODEL * 2);
  uint16_t* WvT    = (uint16_t*)carve((size_t)DMODEL * DMODEL * 2);
  uint16_t* WoT    = (uint16_t*)carve((size_t)DMODEL * DMODEL * 2);
  uint16_t* q_bf   = (uint16_t*)carve((size_t)T_SEQ * DMODEL * 2);
  uint16_t* k_bf   = (uint16_t*)carve((size_t)T_SEQ * DMODEL * 2);
  uint16_t* vT_bf  = (uint16_t*)carve((size_t)DMODEL * T_SEQ * 2);
  float*    xw1    = (float*)carve((size_t)T_SEQ * RANK_ * 4);
  float*    betaF  = (float*)carve((size_t)T_SEQ * NHEAD * 4);
  float*    wlow   = (float*)carve((size_t)T_SEQ * DMODEL * 4);
  uint16_t* w_bf   = (uint16_t*)carve((size_t)T_SEQ * DMODEL * 2);
  uint16_t* bw_bf  = (uint16_t*)carve((size_t)T_SEQ * DMODEL * 2);
  uint16_t* bwT_bf = (uint16_t*)carve((size_t)DMODEL * T_SEQ * 2);
  uint16_t* TinvT  = (uint16_t*)carve((size_t)NHEAD * NCHUNK * 64 * 64 * 2);
  uint16_t* LkT    = (uint16_t*)carve((size_t)NHEAD * NCHUNK * 64 * 64 * 2);
  float*    AccF   = (float*)carve((size_t)NHEAD * T_SEQ * HDIM * 4);
  float*    Sbuf   = (float*)carve((size_t)NHEAD * T_SEQ * T_SEQ * 4);
  uint16_t* P_bf   = (uint16_t*)carve((size_t)NHEAD * T_SEQ * T_SEQ * 2);
  uint16_t* attn_bf= (uint16_t*)carve((size_t)T_SEQ * DMODEL * 2);

  const int nTD = T_SEQ * DMODEL;
  cast_bf16_kernel<<<nTD / 256, 256, 0, stream>>>(x, x_bf, nTD);
  transpose_cast_kernel<<<nTD / 256, 256, 0, stream>>>(Wq, WqT, DMODEL, DMODEL);
  transpose_cast_kernel<<<nTD / 256, 256, 0, stream>>>(Wk, WkT, DMODEL, DMODEL);
  transpose_cast_kernel<<<nTD / 256, 256, 0, stream>>>(Wv, WvT, DMODEL, DMODEL);
  transpose_cast_kernel<<<nTD / 256, 256, 0, stream>>>(Wo, WoT, DMODEL, DMODEL);

  proj_beta_kernel<<<(T_SEQ * NHEAD) / 256, 256, 0, stream>>>(x, Wb, betaF);
  proj_xw1_kernel<<<(T_SEQ * RANK_) / 256, 256, 0, stream>>>(x, Ww1, xw1);
  wlow_kernel<<<nTD / 256, 256, 0, stream>>>(xw1, Ww2, wlow);
  conv_norm_kernel<<<dim3(T_SEQ, NHEAD), HDIM, 0, stream>>>(wlow, convw, betaF, w_bf, bw_bf, bwT_bf);

  const dim3 g16(DMODEL / 64, T_SEQ / 64, 1);
  gemm64_kernel<1><<<g16, 32, 0, stream>>>(x_bf, DMODEL, 0, WqT, DMODEL, 0, q_bf, DMODEL, 0, DMODEL, 1.f);
  gemm64_kernel<1><<<g16, 32, 0, stream>>>(x_bf, DMODEL, 0, WkT, DMODEL, 0, k_bf, DMODEL, 0, DMODEL, 1.f);
  gemm64_kernel<2><<<g16, 32, 0, stream>>>(x_bf, DMODEL, 0, WvT, DMODEL, 0, vT_bf, T_SEQ, 0, DMODEL, 1.f);

  invert_kernel<<<dim3(NCHUNK, NHEAD), 64, 0, stream>>>(w_bf, k_bf, bw_bf, TinvT, LkT);

  hipMemsetAsync(AccF, 0, (size_t)NHEAD * T_SEQ * HDIM * sizeof(float), stream);
  for (int J = NCHUNK - 1; J >= 0; --J)
    path_step_kernel<<<dim3(NCHUNK - J, NHEAD), 32, 0, stream>>>(
        J, q_bf, k_bf, w_bf, bwT_bf, TinvT, LkT, AccF, Sbuf);

  softmax_kernel<<<dim3(T_SEQ, NHEAD), 256, 0, stream>>>(Sbuf, P_bf);

  // o_h = P_h @ v_h : batched over heads (A = P row-major, B^T = v^T rows)
  gemm64_kernel<1><<<dim3(1, T_SEQ / 64, NHEAD), 32, 0, stream>>>(
      P_bf, T_SEQ, (long long)T_SEQ * T_SEQ,
      vT_bf, T_SEQ, (long long)HDIM * T_SEQ,
      attn_bf, DMODEL, (long long)HDIM, T_SEQ, 1.f);

  // out = attn @ Wo (fp32 result)
  gemm64_kernel<0><<<g16, 32, 0, stream>>>(attn_bf, DMODEL, 0, WoT, DMODEL, 0, out, DMODEL, 0, DMODEL, 1.f);
}
